// ConvTranspose2dManual_77695958385100
// MI455X (gfx1250) — matrix operand order repaired and checked by reading, not compiled, not run on hardware
//
#include <hip/hip_runtime.h>
#include <stdint.h>

// ConvTranspose2d(k=s=2) == GEMM (M=1024, N=32768, K=512) + pixel shuffle.
// Pre-kernel: weight (512,256,2,2) f32 -> Wt[m=(o,i,j)][k=c] bf16 in d_ws.
// Main kernel: bf16 WMMA GEMM, f32 accumulate, double-buffered LDS.
//  - A tile (bf16 memcpy) staged with global_load_async_to_lds_b128 (ASYNCcnt)
//  - B tile (x, f32) loaded coalesced, packed-converted to bf16, transposed in regs
//  - grid ordered so the 8 m-blocks sharing an x-tile are adjacent (L2 reuse)

#define GK 512      // K  = IN_C
#define GM 1024     // M  = OUT_C*2*2
#define GN 32768    // N  = 8*64*64
#define BM 128
#define BN 128
#define BK 32
#define LDT 40      // LDS row stride in bf16 elements (80 B: 16B-aligned, conflict-free)
#define KSTEPS (GK / BK)

typedef __attribute__((ext_vector_type(16))) __bf16 v16bf;
typedef __attribute__((ext_vector_type(4)))  __bf16 v4bf;
typedef __attribute__((ext_vector_type(4)))  float  v4f;
typedef __attribute__((ext_vector_type(8)))  float  v8f;

// ---- weight transpose + convert: w[k][m] f32 -> wt[m][k] bf16 --------------
__global__ __launch_bounds__(256) void wt_transpose_bf16(
    const float* __restrict__ w, __bf16* __restrict__ wt) {
  __shared__ float tile[32][33];
  const int bk   = blockIdx.x & 15;   // 512/32  k-tiles
  const int bm   = blockIdx.x >> 4;   // 1024/32 m-tiles
  const int col  = threadIdx.x & 31;
  const int row0 = threadIdx.x >> 5;  // 0..7
#pragma unroll
  for (int p = 0; p < 4; ++p) {
    const int r = row0 + p * 8;
    tile[r][col] = w[(size_t)(bk * 32 + r) * GM + bm * 32 + col];
  }
  __syncthreads();
#pragma unroll
  for (int p = 0; p < 4; ++p) {
    const int r = row0 + p * 8;
    wt[(size_t)(bm * 32 + r) * GK + bk * 32 + col] = (__bf16)tile[col][r];
  }
}

// ---- main GEMM -------------------------------------------------------------
__global__ __launch_bounds__(256) void dconv_gemm_wmma(
    const float* __restrict__ x, const float* __restrict__ bias,
    const __bf16* __restrict__ wt, float* __restrict__ out) {
  __shared__ __bf16 As[2][BM * LDT];   // [m][k] bf16
  __shared__ __bf16 Bs[2][BN * LDT];   // [n][k] bf16

  const int t    = threadIdx.x;
  const int lane = t & 31;
  const int lo16 = (lane < 16);
  const int l15  = lane & 15;
  const int wv   = t >> 5;
  const int wr   = (wv >> 2) * 64;   // wave m-offset (2 rows of waves)
  const int wc   = (wv & 3) * 32;    // wave n-offset (4 cols of waves)

  const int m0 = blockIdx.x * BM;    // feature base (fast grid axis: 8 blocks
  const int n0 = blockIdx.y * BN;    //   sharing one x-tile run back-to-back)
  const int b  = n0 >> 12;
  const int s0 = n0 & 4095;

  const float* xblk = x + (size_t)b * GK * 4096 + s0;

  // A staging: 128 rows x 64B = 256 threads x 2 async 16B chunks
  const int arow = t >> 2;           // 0..63
  const int aq   = (t & 3) * 8;      // element offset (16B quarters)
  // B staging: thread loads 4 k-rows x 4 n (float4), converts+transposes in regs
  const int kq = (t & 7) * 4;        // 0,4,...,28
  const int nn = (t >> 3) * 4;       // 0..124

  float4 bR[4];

  auto asyncCopyA = [&](int buf, int k0) {
    const __bf16* g0 = wt + (size_t)(m0 + arow) * GK + k0 + aq;
    const __bf16* g1 = wt + (size_t)(m0 + arow + 64) * GK + k0 + aq;
    const uint32_t l0 = (uint32_t)(uintptr_t)&As[buf][arow * LDT + aq];
    const uint32_t l1 = (uint32_t)(uintptr_t)&As[buf][(arow + 64) * LDT + aq];
    asm volatile("global_load_async_to_lds_b128 %0, %1, off"
                 :: "v"(l0), "v"(g0) : "memory");
    asm volatile("global_load_async_to_lds_b128 %0, %1, off"
                 :: "v"(l1), "v"(g1) : "memory");
  };
  auto loadRegsB = [&](int k0) {
#pragma unroll
    for (int dk = 0; dk < 4; ++dk)
      bR[dk] = *(const float4*)(xblk + (size_t)(k0 + kq + dk) * 4096 + nn);
  };
  auto storeStageB = [&](int buf) {
    const float* bf = (const float*)&bR[0];
#pragma unroll
    for (int dn = 0; dn < 4; ++dn) {
      v4f fv = {bf[0 * 4 + dn], bf[1 * 4 + dn], bf[2 * 4 + dn], bf[3 * 4 + dn]};
      *(v4bf*)&Bs[buf][(nn + dn) * LDT + kq] =
          __builtin_convertvector(fv, v4bf);   // -> v_cvt_pk_bf16_f32
    }
  };

  v8f acc[4][2];
#pragma unroll
  for (int a = 0; a < 4; ++a)
#pragma unroll
    for (int c = 0; c < 2; ++c)
      acc[a][c] = v8f{0.f, 0.f, 0.f, 0.f, 0.f, 0.f, 0.f, 0.f};

  asyncCopyA(0, 0);
  loadRegsB(0);
  storeStageB(0);
  asm volatile("s_wait_asynccnt 0x0" ::: "memory");
  __syncthreads();

  for (int ks = 0; ks < KSTEPS; ++ks) {
    const int buf = ks & 1;
    if (ks + 1 < KSTEPS) {
      asyncCopyA(buf ^ 1, (ks + 1) * BK);   // buffer released by prior barrier
      loadRegsB((ks + 1) * BK);
    }

    union Frag { uint4 q[2]; v16bf v; };
    Frag fa[4], fb[2];
#pragma unroll
    for (int a = 0; a < 4; ++a) {
      // A 16x32 bf16 layout: lane<16 -> K 0..7 & 16..23; lane>=16 -> K 8..15 & 24..31
      const __bf16* p = &As[buf][(wr + a * 16 + l15) * LDT + (lo16 ? 0 : 8)];
      fa[a].q[0] = *(const uint4*)p;
      fa[a].q[1] = *(const uint4*)(p + 16);
    }
#pragma unroll
    for (int c = 0; c < 2; ++c) {
      // B 32x16 bf16 layout: lane<16 -> N=lane, K 0..7 & 16..23; lane>=16 -> K 8..15 & 24..31
      const __bf16* p = &Bs[buf][(wc + c * 16 + l15) * LDT + (lo16 ? 0 : 8)];
      fb[c].q[0] = *(const uint4*)p;
      fb[c].q[1] = *(const uint4*)(p + 16);
    }
#pragma unroll
    for (int a = 0; a < 4; ++a)
#pragma unroll
      for (int c = 0; c < 2; ++c)
        acc[a][c] = __builtin_amdgcn_wmma_f32_16x16x32_bf16(
            false, fa[a].v, false, fb[c].v, (short)0, acc[a][c], false, false);

    if (ks + 1 < KSTEPS) {
      storeStageB(buf ^ 1);
      asm volatile("s_wait_asynccnt 0x0" ::: "memory");
      __syncthreads();
    }
  }

  // ---- epilogue: pixel-shuffle scatter + bias ----
  const int mhalf = (lane & 16) >> 1;   // lanes>=16 hold M rows +8
#pragma unroll
  for (int a = 0; a < 4; ++a) {
#pragma unroll
    for (int r = 0; r < 8; ++r) {
      const int m = m0 + wr + a * 16 + mhalf + r;   // m = o*4 + i*2 + j
      const int o = m >> 2, i = (m >> 1) & 1, j = m & 1;
      const float bv = bias[o];
      const size_t obase = ((size_t)b * 256 + o) * 16384;
#pragma unroll
      for (int c = 0; c < 2; ++c) {
        const int n  = n0 + wc + c * 16 + l15;      // n = b*4096 + h*64 + w
        const int hw = n & 4095;
        const int h = hw >> 6, wcol = hw & 63;
        out[obase + (size_t)(2 * h + i) * 128 + (2 * wcol + j)] = acc[a][c][r] + bv;
      }
    }
  }
}

extern "C" void kernel_launch(void* const* d_in, const int* in_sizes, int n_in,
                              void* d_out, int out_size, void* d_ws, size_t ws_size,
                              hipStream_t stream) {
  const float* x    = (const float*)d_in[0];   // (8,512,64,64) f32
  const float* w    = (const float*)d_in[1];   // (512,256,2,2) f32
  const float* bias = (const float*)d_in[2];   // (256,) f32
  float* out = (float*)d_out;                  // (8,256,128,128) f32
  __bf16* wt = (__bf16*)d_ws;                  // 1024*512 bf16 = 1 MB scratch

  wt_transpose_bf16<<<512, 256, 0, stream>>>(w, wt);
  dconv_gemm_wmma<<<dim3(GM / BM, GN / BN), 256, 0, stream>>>(x, bias, wt, out);
}
